// GATClassifier_5188320494191
// MI455X (gfx1250) — compile-verified, matrix operand order checked by reading, not run on hardware
//
#include <hip/hip_runtime.h>
#include <math.h>

typedef __attribute__((ext_vector_type(16))) __bf16 v16bf;
typedef __attribute__((ext_vector_type(8)))  __bf16 v8bf;
typedef __attribute__((ext_vector_type(8)))  float  v8f;

#define N_NODES   20000
#define N_EDGES   160000
#define E_TOT     (N_EDGES + N_NODES)
#define EPS_BN    1e-5f
#define NEG_SLOPE 0.2f

// ---------------- atomic float max (ordered-int trick) ----------------
__device__ __forceinline__ void atomicMaxF(float* addr, float val) {
  if (val >= 0.0f) atomicMax((int*)addr, __float_as_int(val));
  else             atomicMin((unsigned int*)addr, __float_as_uint(val));
}

__device__ __forceinline__ float eluf(float x) { return x > 0.0f ? x : expm1f(x); }

// ---------------- convert + transpose weights: Wt[n*K+k] = bf16(W[k*N+n]) ----------------
__global__ void k_transpose_bf16(const float* __restrict__ W, __bf16* __restrict__ Wt,
                                 int K, int N) {
  int tid = blockIdx.x * blockDim.x + threadIdx.x;
  if (tid >= K * N) return;
  int n = tid / K, k = tid - n * K;
  Wt[(size_t)n * K + k] = (__bf16)W[(size_t)k * N + n];
}

// ---------------- WMMA GEMM: C[M,N] = A[M,K] (f32, row-major, converted on the fly)
//                               * Bt[N,K] (bf16, pre-transposed)
// block = 256 threads = 8 waves; wave tile = 32 rows x 64 cols (2 A frags x 4 B frags);
// block tile = 256 x 64. Each B fragment feeds 2 WMMAs -> 32 B loaded per WMMA per lane.
__global__ void k_wmma_gemm(const float* __restrict__ A, const __bf16* __restrict__ Bt,
                            float* __restrict__ C, int M, int N, int K) {
  const int wave = threadIdx.x >> 5;
  const int lane = threadIdx.x & 31;
  const int hi   = lane >> 4;     // half-wave select (ISA 16-bit A layout)
  const int lq   = lane & 15;
  const int rowTile = blockIdx.y * 256 + wave * 32;
  const int n0 = blockIdx.x * 64;

  int arow0 = rowTile + lq;
  int arow1 = rowTile + 16 + lq;
  if (arow0 >= M) arow0 = M - 1;  // keep EXEC all-1s; stores are guarded
  if (arow1 >= M) arow1 = M - 1;

  v8f acc0[4] = {};
  v8f acc1[4] = {};

  for (int k = 0; k < K; k += 32) {
    // A fragments: lane holds K = k + hi*8 + {0..7} and k + 16 + hi*8 + {0..7}
    const float* ap0 = A + (size_t)arow0 * K + k + hi * 8;
    const float* ap1 = A + (size_t)arow1 * K + k + hi * 8;
    v8f a0lo = *(const v8f*)(ap0);
    v8f a0hi = *(const v8f*)(ap0 + 16);
    v8f a1lo = *(const v8f*)(ap1);
    v8f a1hi = *(const v8f*)(ap1 + 16);
    v16bf a0, a1;
#pragma unroll
    for (int e = 0; e < 8; ++e) {
      a0[e] = (__bf16)a0lo[e]; a0[e + 8] = (__bf16)a0hi[e];
      a1[e] = (__bf16)a1lo[e]; a1[e + 8] = (__bf16)a1hi[e];
    }

#pragma unroll
    for (int t = 0; t < 4; ++t) {
      const __bf16* bp = Bt + (size_t)(n0 + t * 16 + lq) * K + k + hi * 8;
      v8bf blo = *(const v8bf*)(bp);
      v8bf bhi = *(const v8bf*)(bp + 16);
      v16bf b = __builtin_shufflevector(blo, bhi, 0,1,2,3,4,5,6,7,8,9,10,11,12,13,14,15);
      acc0[t] = __builtin_amdgcn_wmma_f32_16x16x32_bf16(
          false, a0, false, b, (short)0, acc0[t], false, false);
      acc1[t] = __builtin_amdgcn_wmma_f32_16x16x32_bf16(
          false, a1, false, b, (short)0, acc1[t], false, false);
    }
  }

  // C/D layout: VGPR r -> row = r + hi*8, col = lane&15
#pragma unroll
  for (int t = 0; t < 4; ++t) {
#pragma unroll
    for (int r = 0; r < 8; ++r) {
      int m0 = rowTile + hi * 8 + r;
      int m1 = m0 + 16;
      if (m0 < M) C[(size_t)m0 * N + n0 + t * 16 + lq] = acc0[t][r];
      if (m1 < M) C[(size_t)m1 * N + n0 + t * 16 + lq] = acc1[t][r];
    }
  }
}

// ------- per-node attention logits, one wave per (n,h): coalesced + shfl reduce -------
__global__ void k_att(const float* __restrict__ h, const float* __restrict__ asrc,
                      const float* __restrict__ adst, float* __restrict__ out_s,
                      float* __restrict__ out_d, int H, int C) {
  int gw   = (blockIdx.x * blockDim.x + threadIdx.x) >> 5;   // wave index = n*H + hh
  int lane = threadIdx.x & 31;
  if (gw >= N_NODES * H) return;
  int hh = gw % H;
  const float* hp = h + (size_t)gw * C;
  float ss = 0.0f, sd = 0.0f;
  for (int c = lane; c < C; c += 32) {
    float v = hp[c];
    ss += v * asrc[hh * C + c];
    sd += v * adst[hh * C + c];
  }
#pragma unroll
  for (int o = 16; o > 0; o >>= 1) {
    ss += __shfl_xor(ss, o, 32);
    sd += __shfl_xor(sd, o, 32);
  }
  if (lane == 0) { out_s[gw] = ss; out_d[gw] = sd; }
}

__global__ void k_init_md(float* __restrict__ m, float* __restrict__ d, int n) {
  int tid = blockIdx.x * blockDim.x + threadIdx.x;
  if (tid < n) { m[tid] = -INFINITY; d[tid] = 0.0f; }
}

__global__ void k_zero(float* __restrict__ p, long long n) {
  long long tid = (long long)blockIdx.x * blockDim.x + threadIdx.x;
  if (tid < n) p[tid] = 0.0f;
}

__device__ __forceinline__ void edge_sd(const int* __restrict__ ei, int eid, int& s, int& d) {
  if (eid < N_EDGES) { s = ei[eid]; d = ei[N_EDGES + eid]; }
  else               { s = d = eid - N_EDGES; }    // self loops
}

// leaky_relu(a_src[src] + a_dst[dst]); segment-max into m[dst]
__global__ void k_edge_logits(const int* __restrict__ ei, const float* __restrict__ a_s,
                              const float* __restrict__ a_d, float* __restrict__ e,
                              float* __restrict__ m, int H) {
  int tid = blockIdx.x * blockDim.x + threadIdx.x;
  if (tid >= E_TOT * H) return;
  int eid = tid / H, hh = tid - eid * H, s, d;
  edge_sd(ei, eid, s, d);
  float v = a_s[s * H + hh] + a_d[d * H + hh];
  v = v > 0.0f ? v : NEG_SLOPE * v;
  e[tid] = v;
  atomicMaxF(&m[d * H + hh], v);
}

// e = exp(e - m[dst]); segment-sum into denom[dst]
__global__ void k_edge_exp(const int* __restrict__ ei, float* __restrict__ e,
                           const float* __restrict__ m, float* __restrict__ denom, int H) {
  int tid = blockIdx.x * blockDim.x + threadIdx.x;
  if (tid >= E_TOT * H) return;
  int eid = tid / H, hh = tid - eid * H, s, d;
  edge_sd(ei, eid, s, d);
  float v = expf(e[tid] - m[d * H + hh]);
  e[tid] = v;
  atomicAdd(&denom[d * H + hh], v);
}

// agg[dst,h,:] += alpha * h[src,h,:]; one wave per (edge,head) -> coalesced atomics
__global__ void k_aggregate(const int* __restrict__ ei, const float* __restrict__ e,
                            const float* __restrict__ denom, const float* __restrict__ h,
                            float* __restrict__ agg, int H, int C) {
  long long t = (long long)blockIdx.x * blockDim.x + threadIdx.x;
  int gw   = (int)(t >> 5);                 // wave index = eid*H + hh
  int lane = threadIdx.x & 31;
  if (gw >= E_TOT * H) return;
  int eid = gw / H, hh = gw - eid * H, s, d;
  edge_sd(ei, eid, s, d);
  float alpha = e[gw] / (denom[d * H + hh] + 1e-16f);
  const float* hp = h + ((size_t)s * H + hh) * C;
  float* op = agg + ((size_t)d * H + hh) * C;
  for (int c = lane; c < C; c += 32) atomicAdd(&op[c], alpha * hp[c]);
}

// per-channel mean/var over rows (biased var), input = x + bias (broadcast)
// block = 256 = 32 cols x 8 row-stripes -> coalesced 128B reads per wave
__global__ void k_bn_stats(const float* __restrict__ x, const float* __restrict__ bias,
                           float* __restrict__ mean, float* __restrict__ var,
                           int rows, int cols) {
  __shared__ float sh[8][33], sh2[8][33];
  int tx = threadIdx.x & 31, ty = threadIdx.x >> 5;
  int col = blockIdx.x * 32 + tx;          // cols is a multiple of 32
  float bcol = bias[col];
  float s = 0.0f, s2 = 0.0f;
  for (int r = ty; r < rows; r += 8) {
    float v = x[(size_t)r * cols + col] + bcol;
    s += v; s2 += v * v;
  }
  sh[ty][tx] = s; sh2[ty][tx] = s2;
  __syncthreads();
  if (ty == 0) {
#pragma unroll
    for (int i = 1; i < 8; ++i) { s += sh[i][tx]; s2 += sh2[i][tx]; }
    float mu = s / rows;
    mean[col] = mu;
    var[col] = s2 / rows - mu * mu;
  }
}

__global__ void k_bn_elu(const float* __restrict__ x, const float* __restrict__ bias,
                         const float* __restrict__ mean, const float* __restrict__ var,
                         const float* __restrict__ g, const float* __restrict__ b,
                         float* __restrict__ y, int rows, int cols) {
  long long tid = (long long)blockIdx.x * blockDim.x + threadIdx.x;
  if (tid >= (long long)rows * cols) return;
  int col = (int)(tid % cols);
  float v = x[tid] + bias[col];
  v = (v - mean[col]) * rsqrtf(var[col] + EPS_BN) * g[col] + b[col];
  y[tid] = eluf(v);
}

// Y[n,j] = elu(sum_k X[n,k]*W[k,J] + b[j])
__global__ void k_fc_elu(const float* __restrict__ X, const float* __restrict__ W,
                         const float* __restrict__ b, float* __restrict__ Y,
                         int rows, int K, int J) {
  int tid = blockIdx.x * blockDim.x + threadIdx.x;
  if (tid >= rows * J) return;
  int n = tid / J, j = tid - n * J;
  const float* xp = X + (size_t)n * K;
  float s = b[j];
  for (int k = 0; k < K; ++k) s += xp[k] * W[k * J + j];
  Y[tid] = eluf(s);
}

// out[n,:2] = z1[n,:] @ Wc2 + bc2
__global__ void k_fc_out(const float* __restrict__ Z, const float* __restrict__ W,
                         const float* __restrict__ b, float* __restrict__ out, int rows) {
  int n = blockIdx.x * blockDim.x + threadIdx.x;
  if (n >= rows) return;
  const float* zp = Z + (size_t)n * 64;
  float s0 = b[0], s1 = b[1];
  for (int j = 0; j < 64; ++j) {
    float v = zp[j];
    s0 += v * W[j * 2 + 0];
    s1 += v * W[j * 2 + 1];
  }
  out[n * 2 + 0] = s0;
  out[n * 2 + 1] = s1;
}

extern "C" void kernel_launch(void* const* d_in, const int* in_sizes, int n_in,
                              void* d_out, int out_size, void* d_ws, size_t ws_size,
                              hipStream_t stream) {
  const float* x        = (const float*)d_in[0];
  const int*   ei       = (const int*)d_in[1];
  const float* W1       = (const float*)d_in[2];
  const float* att_src1 = (const float*)d_in[3];
  const float* att_dst1 = (const float*)d_in[4];
  const float* bias1    = (const float*)d_in[5];
  const float* bn1_g    = (const float*)d_in[6];
  const float* bn1_b    = (const float*)d_in[7];
  const float* W2       = (const float*)d_in[8];
  const float* att_src2 = (const float*)d_in[9];
  const float* att_dst2 = (const float*)d_in[10];
  const float* bias2    = (const float*)d_in[11];
  const float* bn2_g    = (const float*)d_in[12];
  const float* bn2_b    = (const float*)d_in[13];
  const float* Wc1      = (const float*)d_in[14];
  const float* bc1      = (const float*)d_in[15];
  const float* Wc2      = (const float*)d_in[16];
  const float* bc2      = (const float*)d_in[17];
  float* out = (float*)d_out;

  // ---- workspace carve (256-byte aligned) ----
  char* ws = (char*)d_ws;
  size_t off = 0;
  auto carve = [&](size_t bytes) -> char* {
    char* p = ws + off;
    off += (bytes + 255) & ~(size_t)255;
    return p;
  };
  __bf16* wt   = (__bf16*)carve((size_t)1024 * 768 * 2);          // max of both weight mats
  float* Hbuf  = (float*)carve((size_t)N_NODES * 1024 * 4);       // h1 raw -> h1bn -> h2bn
  float* Gbuf  = (float*)carve((size_t)N_NODES * 1024 * 4);       // agg1 -> {h2raw, agg2, z1}
  float* ebuf  = (float*)carve((size_t)E_TOT * 8 * 4);            // edge logits / alphas
  float* a_s   = (float*)carve((size_t)N_NODES * 8 * 4);
  float* a_d   = (float*)carve((size_t)N_NODES * 8 * 4);
  float* mbuf  = (float*)carve((size_t)N_NODES * 8 * 4);
  float* dbuf  = (float*)carve((size_t)N_NODES * 8 * 4);
  float* meanb = (float*)carve(1024 * 4);
  float* varb  = (float*)carve(1024 * 4);
  (void)ws_size; (void)in_sizes; (void)n_in; (void)out_size;

  dim3 blk(256);
  const int gridMy = (N_NODES + 255) / 256;   // 256-row block tiles

  // ================= Layer 1: GATConv(768 -> 8 heads x 128, concat) =================
  k_transpose_bf16<<<(768 * 1024 + 255) / 256, blk, 0, stream>>>(W1, wt, 768, 1024);
  k_wmma_gemm<<<dim3(1024 / 64, gridMy), blk, 0, stream>>>(x, wt, Hbuf, N_NODES, 1024, 768);
  k_att<<<(N_NODES * 8 * 32 + 255) / 256, blk, 0, stream>>>(Hbuf, att_src1, att_dst1, a_s, a_d, 8, 128);
  k_init_md<<<(N_NODES * 8 + 255) / 256, blk, 0, stream>>>(mbuf, dbuf, N_NODES * 8);
  k_zero<<<(int)(((long long)N_NODES * 1024 + 255) / 256), blk, 0, stream>>>(Gbuf, (long long)N_NODES * 1024);
  k_edge_logits<<<(E_TOT * 8 + 255) / 256, blk, 0, stream>>>(ei, a_s, a_d, ebuf, mbuf, 8);
  k_edge_exp<<<(E_TOT * 8 + 255) / 256, blk, 0, stream>>>(ei, ebuf, mbuf, dbuf, 8);
  k_aggregate<<<(int)(((long long)E_TOT * 8 * 32 + 255) / 256), blk, 0, stream>>>(
      ei, ebuf, dbuf, Hbuf, Gbuf, 8, 128);
  k_bn_stats<<<1024 / 32, blk, 0, stream>>>(Gbuf, bias1, meanb, varb, N_NODES, 1024);
  k_bn_elu<<<(int)(((long long)N_NODES * 1024 + 255) / 256), blk, 0, stream>>>(
      Gbuf, bias1, meanb, varb, bn1_g, bn1_b, Hbuf, N_NODES, 1024);

  // ================= Layer 2: GATConv(1024 -> 1 head x 128) =================
  float* h2   = Gbuf;                                  // [N,128]  (agg1 is dead now)
  float* agg2 = Gbuf + (size_t)N_NODES * 128;          // [N,128]
  float* z1   = Gbuf + (size_t)2 * N_NODES * 128;      // [N,64]
  k_transpose_bf16<<<(1024 * 128 + 255) / 256, blk, 0, stream>>>(W2, wt, 1024, 128);
  k_wmma_gemm<<<dim3(128 / 64, gridMy), blk, 0, stream>>>(Hbuf, wt, h2, N_NODES, 128, 1024);
  k_att<<<(N_NODES * 32 + 255) / 256, blk, 0, stream>>>(h2, att_src2, att_dst2, a_s, a_d, 1, 128);
  k_init_md<<<(N_NODES + 255) / 256, blk, 0, stream>>>(mbuf, dbuf, N_NODES);
  k_zero<<<(int)(((long long)N_NODES * 128 + 255) / 256), blk, 0, stream>>>(agg2, (long long)N_NODES * 128);
  k_edge_logits<<<(E_TOT + 255) / 256, blk, 0, stream>>>(ei, a_s, a_d, ebuf, mbuf, 1);
  k_edge_exp<<<(E_TOT + 255) / 256, blk, 0, stream>>>(ei, ebuf, mbuf, dbuf, 1);
  k_aggregate<<<(int)(((long long)E_TOT * 32 + 255) / 256), blk, 0, stream>>>(
      ei, ebuf, dbuf, h2, agg2, 1, 128);
  k_bn_stats<<<128 / 32, blk, 0, stream>>>(agg2, bias2, meanb, varb, N_NODES, 128);
  k_bn_elu<<<(int)(((long long)N_NODES * 128 + 255) / 256), blk, 0, stream>>>(
      agg2, bias2, meanb, varb, bn2_g, bn2_b, Hbuf, N_NODES, 128);

  // ================= Classifier =================
  k_fc_elu<<<(N_NODES * 64 + 255) / 256, blk, 0, stream>>>(Hbuf, Wc1, bc1, z1, N_NODES, 128, 64);
  k_fc_out<<<(N_NODES + 255) / 256, blk, 0, stream>>>(z1, Wc2, bc2, out, N_NODES);
}
